// SparseResBlock_58935541236229
// MI455X (gfx1250) — matrix operand order repaired and checked by reading, not compiled, not run on hardware
//
#include <hip/hip_runtime.h>

typedef __attribute__((ext_vector_type(16))) _Float16 v16h;
typedef __attribute__((ext_vector_type(8)))  _Float16 v8h;
typedef __attribute__((ext_vector_type(8)))  float    v8f;
typedef __attribute__((ext_vector_type(4)))  float    v4f;
typedef __attribute__((ext_vector_type(4)))  unsigned int v4u;
typedef __attribute__((ext_vector_type(8)))  int      v8i;
typedef __attribute__((ext_vector_type(4)))  int      v4i;

#define C64  64
#define KVOL 27
#define WROWS (KVOL * C64)        // 1728 weight rows (k,cout)
#define WPITCH 72                 // LDS row pitch in halves (128B row + 16B pad)
#define WLDS_BYTES (WROWS * WPITCH * 2)  // 248832 bytes of dynamic LDS

// ---------------------------------------------------------------------------
// Small utility kernels
// ---------------------------------------------------------------------------

__global__ void zero_stats_kernel(float* __restrict__ stats) {
  int i = blockIdx.x * blockDim.x + threadIdx.x;
  if (i < 512) stats[i] = 0.0f;
}

__global__ void cvt_feat_kernel(const float* __restrict__ f,
                                _Float16* __restrict__ h, size_t n4) {
  size_t i = (size_t)blockIdx.x * blockDim.x + threadIdx.x;
  if (i >= n4) return;
  size_t base = i * 4;
  v4f v = *(const v4f*)(f + base);
  h[base + 0] = (_Float16)v.x;
  h[base + 1] = (_Float16)v.y;
  h[base + 2] = (_Float16)v.z;
  h[base + 3] = (_Float16)v.w;
}

// W[k][cin][cout] (f32)  ->  Wt[k][cout][cin] (f16)
__global__ void cvt_w_kernel(const float* __restrict__ W1,
                             const float* __restrict__ W2,
                             _Float16* __restrict__ T1,
                             _Float16* __restrict__ T2, int kcc) {
  int i = blockIdx.x * blockDim.x + threadIdx.x;
  if (i >= 2 * kcc) return;
  const float* W = (i < kcc) ? W1 : W2;
  _Float16*    T = (i < kcc) ? T1 : T2;
  int j  = (i < kcc) ? i : i - kcc;
  int k  = j / (C64 * C64);
  int r  = j % (C64 * C64);
  int co = r / C64;
  int ci = r % C64;
  T[j] = (_Float16)W[k * C64 * C64 + ci * C64 + co];
}

__global__ void bn_finalize_kernel(const float* __restrict__ sum,
                                   const float* __restrict__ ssq,
                                   const float* __restrict__ gamma,
                                   const float* __restrict__ beta,
                                   float* __restrict__ scale,
                                   float* __restrict__ shift,
                                   float n, float eps) {
  int c = threadIdx.x;
  if (c < C64) {
    float mean = sum[c] / n;
    float var  = ssq[c] / n - mean * mean;
    float rstd = rsqrtf(var + eps);
    float g    = gamma[c] * rstd;
    scale[c] = g;
    shift[c] = beta[c] - mean * g;
  }
}

// In-place BN + ReLU on f16 activations (8 contiguous channels per thread).
__global__ void bn_relu_f16_kernel(_Float16* __restrict__ y,
                                   const float* __restrict__ scale,
                                   const float* __restrict__ shift,
                                   size_t total8) {
  size_t i = (size_t)blockIdx.x * blockDim.x + threadIdx.x;
  if (i >= total8) return;
  size_t base = i * 8;
  int c0 = (int)(base & 63);
  v8h v = *(v8h*)(y + base);
  v8h o;
#pragma unroll
  for (int j = 0; j < 8; ++j) {
    float f = (float)v[j] * scale[c0 + j] + shift[c0 + j];
    o[j] = (_Float16)fmaxf(f, 0.0f);
  }
  *(v8h*)(y + base) = o;
}

// In-place final: out = relu(out*scale + shift + feat)
__global__ void bn_res_relu_kernel(float* __restrict__ y,
                                   const float* __restrict__ feat,
                                   const float* __restrict__ scale,
                                   const float* __restrict__ shift,
                                   size_t total4) {
  size_t i = (size_t)blockIdx.x * blockDim.x + threadIdx.x;
  if (i >= total4) return;
  size_t base = i * 4;
  int c0 = (int)(base & 63);
  v4f v = *(v4f*)(y + base);
  v4f f = *(const v4f*)(feat + base);
#pragma unroll
  for (int j = 0; j < 4; ++j)
    v[j] = fmaxf(v[j] * scale[c0 + j] + shift[c0 + j] + f[j], 0.0f);
  *(v4f*)(y + base) = v;
}

// ---------------------------------------------------------------------------
// Gather-GEMM submanifold conv via v_wmma_f32_16x16x32_f16.
// Weights are DMA'd once per workgroup into LDS by the Tensor Data Mover
// (2D tile, LDS row padded 128B->144B for bank-conflict-free ds_load_b128),
// then every wave streams 16-voxel blocks: gather A from L2-resident
// activations, B from LDS, 8 WMMAs per kernel offset, fused BN statistics.
// ---------------------------------------------------------------------------
template <typename OUT_T>
__global__ __launch_bounds__(256) void subm_conv_wmma(
    const _Float16* __restrict__ X,    // [N,64] f16 activations
    const _Float16* __restrict__ Wt,   // [27][64cout][64cin] f16 (global)
    const int*      __restrict__ nbr,  // [27,N]
    const float*    __restrict__ valid,// [27,N]
    OUT_T*          __restrict__ Y,    // [N,64]
    float* __restrict__ gsum,          // [64]
    float* __restrict__ gssq,          // [64]
    int N, int nblocks, int totalWaves)
{
  __shared__ float lsum[C64];
  __shared__ float lssq[C64];
  extern __shared__ _Float16 wlds[];   // [1728][72] halves, dynamic LDS

  const int tid = threadIdx.x;
  if (tid < C64) { lsum[tid] = 0.0f; lssq[tid] = 0.0f; }

  // --- TDM fill: global Wt (1728 x 128B rows) -> LDS with +16B row padding.
  if (tid == 0) {
    unsigned long long ga = (unsigned long long)(const void*)Wt;
    v4u g0;
    g0.x = 1u;                                   // count=1 valid descriptor
    g0.y = (unsigned)__builtin_amdgcn_groupstaticsize();  // lds_addr (bytes)
    g0.z = (unsigned)(ga & 0xffffffffu);          // global_addr[31:0]
    g0.w = (unsigned)((ga >> 32) & 0x01ffffffu)   // global_addr[56:32]
         | (2u << 30);                            // type = 2 ("image")
    v8i g1;
    g1[0] = (3 << 16)      // data_size = 8 bytes
          | (1 << 20)      // pad_enable
          | (4 << 22)      // pad_interval: pad after 32 DWORDs (one 128B row)
          | (3 << 25);     // pad_amount: 4 DWORDs (16B) -> 144B LDS pitch
    g1[1] = (int)(16u << 16);          // tensor_dim0 = 16 (8B units, 128B row)
    g1[2] = (int)((unsigned)WROWS << 16);  // tensor_dim1 = 1728 rows
    g1[3] = (int)(16u << 16);          // tile_dim0 = 16
    g1[4] = WROWS;                     // tile_dim1 = 1728
    g1[5] = 16;                        // tensor_dim0_stride = 16 (contiguous)
    g1[6] = 0;
    g1[7] = 0;
    v4i gz = {0, 0, 0, 0};
#if defined(__clang_major__) && __clang_major__ >= 23
    v8i g4 = {0, 0, 0, 0, 0, 0, 0, 0};
    __builtin_amdgcn_tensor_load_to_lds(g0, g1, gz, gz, g4, 0);
#else
    __builtin_amdgcn_tensor_load_to_lds(g0, g1, gz, gz, 0);
#endif
    __builtin_amdgcn_s_wait_tensorcnt(0);
  }
  __syncthreads();

  const int lane = tid & 31;
  const int half = lane >> 4;   // 0: lanes 0-15, 1: lanes 16-31
  const int m    = lane & 15;   // row within 16x16 tile / column index
  const int wave = (blockIdx.x << 3) | (tid >> 5);

  float psum[4] = {0.f, 0.f, 0.f, 0.f};
  float pssq[4] = {0.f, 0.f, 0.f, 0.f};

  for (int b = wave; b < nblocks; b += totalWaves) {
    v8f acc[4] = {};
    const int v    = b * 16 + m;
    const bool vok = (v < N);
    const int  vc  = vok ? v : (N - 1);

    for (int k = 0; k < KVOL; ++k) {
      const int   nb  = nbr[(size_t)k * N + vc];
      const float val = vok ? valid[(size_t)k * N + vc] : 0.0f;
      const bool  on  = (val != 0.0f);

      // --- A tile: 16 voxels x 32 cin, ISA 16-bit A layout:
      //   lanes 0-15 hold K {0..7, 16..23}; lanes 16-31 hold K {8..15, 24..31}
      const v8h* src = (const v8h*)(X + (size_t)nb * C64);
      v8h z8 = {};
      v8h a0l = on ? src[half]     : z8;   // chunk0: cin {0..7}/{8..15}
      v8h a0h = on ? src[half + 2] : z8;   // chunk0: cin {16..23}/{24..31}
      v8h a1l = on ? src[half + 4] : z8;   // chunk1: cin {32..39}/{40..47}
      v8h a1h = on ? src[half + 6] : z8;   // chunk1: cin {48..55}/{56..63}
      v16h A0 = __builtin_shufflevector(a0l, a0h, 0,1,2,3,4,5,6,7,8,9,10,11,12,13,14,15);
      v16h A1 = __builtin_shufflevector(a1l, a1h, 0,1,2,3,4,5,6,7,8,9,10,11,12,13,14,15);

      // --- B tiles from LDS: 32 cin x 16 cout, lanes 0-15 hold
      //   K {0..7, 16..23} of their column, lanes 16-31 hold K {8..15, 24..31}.
#pragma unroll
      for (int t = 0; t < 4; ++t) {
        const _Float16* wp = wlds + (size_t)(k * C64 + t * 16 + m) * WPITCH
                                  + half * 8;
        const v8h* w8 = (const v8h*)wp;
        v16h B0 = __builtin_shufflevector(w8[0], w8[2], 0,1,2,3,4,5,6,7,8,9,10,11,12,13,14,15);
        acc[t] = __builtin_amdgcn_wmma_f32_16x16x32_f16(
            false, A0, false, B0, (short)0, acc[t], false, false);
        const v8h* w8b = (const v8h*)(wp + 32);
        v16h B1 = __builtin_shufflevector(w8b[0], w8b[2], 0,1,2,3,4,5,6,7,8,9,10,11,12,13,14,15);
        acc[t] = __builtin_amdgcn_wmma_f32_16x16x32_f16(
            false, A1, false, B1, (short)0, acc[t], false, false);
      }
    }

    // --- store D tiles + fused BN statistics.
    // D layout: VGPR r, lane: row = r + 8*half, col = m (per ISA C/D table).
#pragma unroll
    for (int t = 0; t < 4; ++t) {
      const int cbase = t * 16 + m;
      float s = 0.0f, q = 0.0f;
#pragma unroll
      for (int r = 0; r < 8; ++r) {
        float x = acc[t][r];
        s += x;
        q += x * x;
        int row = b * 16 + r + half * 8;
        if (row < N) Y[(size_t)row * C64 + cbase] = (OUT_T)x;
      }
      psum[t] += s;
      pssq[t] += q;
    }
  }

#pragma unroll
  for (int t = 0; t < 4; ++t) {
    atomicAdd(&lsum[t * 16 + m], psum[t]);
    atomicAdd(&lssq[t * 16 + m], pssq[t]);
  }
  __syncthreads();
  if (tid < C64) {
    atomicAdd(&gsum[tid], lsum[tid]);
    atomicAdd(&gssq[tid], lssq[tid]);
  }
}

// ---------------------------------------------------------------------------
// Host-side launcher
// ---------------------------------------------------------------------------
extern "C" void kernel_launch(void* const* d_in, const int* in_sizes, int n_in,
                              void* d_out, int out_size, void* d_ws, size_t ws_size,
                              hipStream_t stream) {
  const float* feat  = (const float*)d_in[0];
  const int*   nbr   = (const int*)  d_in[1];
  const float* valid = (const float*)d_in[2];
  const float* W1    = (const float*)d_in[3];
  const float* g1    = (const float*)d_in[4];
  const float* b1    = (const float*)d_in[5];
  const float* W2    = (const float*)d_in[6];
  const float* g2    = (const float*)d_in[7];
  const float* b2    = (const float*)d_in[8];
  float* out = (float*)d_out;

  const int N = in_sizes[0] / C64;
  const int kcc = KVOL * C64 * C64;

  // Workspace layout
  char* ws = (char*)d_ws;
  float*    stats = (float*)ws;                 // 512 floats
  _Float16* Wt1   = (_Float16*)(ws + 2048);
  _Float16* Wt2   = Wt1 + kcc;
  size_t off = 2048 + (size_t)2 * kcc * sizeof(_Float16);
  off = (off + 255) & ~(size_t)255;
  _Float16* xh = (_Float16*)(ws + off);         // [N,64] f16 input features
  _Float16* y1 = xh + (size_t)N * C64;          // [N,64] f16 conv1 output

  float* sum1 = stats;       float* ssq1 = stats + 64;
  float* sc1  = stats + 128; float* sh1  = stats + 192;
  float* sum2 = stats + 256; float* ssq2 = stats + 320;
  float* sc2  = stats + 384; float* sh2  = stats + 448;

  zero_stats_kernel<<<2, 256, 0, stream>>>(stats);

  {
    size_t n4 = (size_t)N * C64 / 4;
    cvt_feat_kernel<<<(unsigned)((n4 + 255) / 256), 256, 0, stream>>>(feat, xh, n4);
  }
  cvt_w_kernel<<<(2 * kcc + 255) / 256, 256, 0, stream>>>(W1, W2, Wt1, Wt2, kcc);

  const int nblocks = (N + 15) / 16;
  const int WGS = 512;
  const int totalWaves = WGS * 8;

  subm_conv_wmma<_Float16><<<WGS, 256, WLDS_BYTES, stream>>>(
      xh, Wt1, nbr, valid, y1, sum1, ssq1, N, nblocks, totalWaves);
  bn_finalize_kernel<<<1, 64, 0, stream>>>(sum1, ssq1, g1, b1, sc1, sh1,
                                           (float)N, 1e-5f);
  {
    size_t n8 = (size_t)N * C64 / 8;
    bn_relu_f16_kernel<<<(unsigned)((n8 + 255) / 256), 256, 0, stream>>>(y1, sc1, sh1, n8);
  }

  subm_conv_wmma<float><<<WGS, 256, WLDS_BYTES, stream>>>(
      y1, Wt2, nbr, valid, out, sum2, ssq2, N, nblocks, totalWaves);
  bn_finalize_kernel<<<1, 64, 0, stream>>>(sum2, ssq2, g2, b2, sc2, sh2,
                                           (float)N, 1e-5f);
  {
    size_t n4 = (size_t)N * C64 / 4;
    bn_res_relu_kernel<<<(unsigned)((n4 + 255) / 256), 256, 0, stream>>>(out, feat, sc2, sh2, n4);
  }
}
